// CrossAttentionBlock_RoPE_83631603187805
// MI455X (gfx1250) — matrix operand order repaired and checked by reading, not compiled, not run on hardware
//
#include <hip/hip_runtime.h>

// ---------------------------------------------------------------------------
// CrossAttentionBlock + RoPE for MI455X (gfx1250), wave32 WMMA pipeline.
// B=8, C=512, H=W=64, NH=8, D=64, window 8x8 (Ws=64), N=4096, NT=32768.
// ---------------------------------------------------------------------------

typedef _Float16 v16h __attribute__((ext_vector_type(16)));
typedef float    v8f  __attribute__((ext_vector_type(8)));

union AF {           // one WMMA f16 operand fragment (16 halves = 8 VGPRs)
    v16h h;
    uint4 u[2];
};
union V32 {          // 32 halves staging for V transpose
    uint4 u[4];
    _Float16 h[32];
};

__device__ __forceinline__ v8f v8f_zero() {
    v8f z = {0.f, 0.f, 0.f, 0.f, 0.f, 0.f, 0.f, 0.f};
    return z;
}

// ---------------------------------------------------------------------------
// LayerNorm: NCHW fp32 -> (token, c) f16 (WMMA A operand)  [+ (b,c,n) fp32]
// one block = 16 tokens, staged through LDS for coalescing both layouts
// ---------------------------------------------------------------------------
__global__ void ln_kernel(const float* __restrict__ x, const float* __restrict__ g,
                          const float* __restrict__ beta, _Float16* __restrict__ out_tc,
                          float* __restrict__ out_ct) {
    __shared__ float sm[512 * 17];
    __shared__ float smu[16], srs[16];
    const int tid  = threadIdx.x;
    const long tok0 = (long)blockIdx.x * 16;
    const int b  = (int)(tok0 >> 12);
    const int n0 = (int)(tok0 & 4095);
    const float* xb = x + (long)b * 512 * 4096;

    for (int idx = tid; idx < 512 * 16; idx += 256) {
        int c = idx >> 4, n = idx & 15;
        sm[c * 17 + n] = xb[(long)c * 4096 + n0 + n];
    }
    __syncthreads();

    const int wv = tid >> 5, lane = tid & 31;
    #pragma unroll
    for (int t = 0; t < 2; ++t) {
        int n = wv * 2 + t;
        float s = 0.f;
        #pragma unroll
        for (int i = 0; i < 16; ++i) s += sm[(lane + 32 * i) * 17 + n];
        #pragma unroll
        for (int m = 1; m < 32; m <<= 1) s += __shfl_xor(s, m, 32);
        float mu = s * (1.f / 512.f);
        float v = 0.f;
        #pragma unroll
        for (int i = 0; i < 16; ++i) {
            float d = sm[(lane + 32 * i) * 17 + n] - mu;
            v += d * d;
        }
        #pragma unroll
        for (int m = 1; m < 32; m <<= 1) v += __shfl_xor(v, m, 32);
        if (lane == 0) {
            smu[n] = mu;
            srs[n] = rsqrtf(v * (1.f / 512.f) + 1e-5f);
        }
    }
    __syncthreads();

    // (token, c) f16 — contiguous in c across threads
    for (int idx = tid; idx < 512 * 16; idx += 256) {
        int n = idx >> 9, c = idx & 511;
        float val = (sm[c * 17 + n] - smu[n]) * srs[n] * g[c] + beta[c];
        out_tc[(tok0 + n) * 512 + c] = (_Float16)val;
    }
    // (b, c, n) fp32 residual copy — contiguous in n across threads
    if (out_ct) {
        float* ob = out_ct + (long)b * 512 * 4096;
        for (int idx = tid; idx < 512 * 16; idx += 256) {
            int c = idx >> 4, n = idx & 15;
            float val = (sm[c * 17 + n] - smu[n]) * srs[n] * g[c] + beta[c];
            ob[(long)c * 4096 + n0 + n] = val;
        }
    }
}

// ---------------------------------------------------------------------------
// Cast the four 512x512 weight matrices to f16 (kept (co, ci) row-major)
// ---------------------------------------------------------------------------
__global__ void wcvt_kernel(const float* __restrict__ a, const float* __restrict__ b,
                            const float* __restrict__ c, const float* __restrict__ d,
                            _Float16* __restrict__ oa, _Float16* __restrict__ ob,
                            _Float16* __restrict__ oc, _Float16* __restrict__ od) {
    int i = blockIdx.x * 256 + threadIdx.x;
    if (i < 512 * 512) {
        oa[i] = (_Float16)a[i];
        ob[i] = (_Float16)b[i];
        oc[i] = (_Float16)c[i];
        od[i] = (_Float16)d[i];
    }
}

// ---------------------------------------------------------------------------
// Expand relative-position bias table -> Bias[h][tq][tk] fp32 (8*64*64)
// ---------------------------------------------------------------------------
__global__ void bias_build_kernel(const float* __restrict__ table, float* __restrict__ Bias) {
    int i = blockIdx.x * 256 + threadIdx.x;
    if (i >= 8 * 64 * 64) return;
    int h  = i >> 12;
    int tq = (i >> 6) & 63;
    int tk = i & 63;
    int dh = (tq >> 3) - (tk >> 3) + 7;
    int dw = (tq & 7) - (tk & 7) + 7;
    Bias[i] = table[(dh * 15 + dw) * 8 + h];
}

// ---------------------------------------------------------------------------
// GEMM: out[m, co] = sum_ci A[m, ci] * W[co, ci] + bias[co]
//   MODE 0: f16 (token,c) output           (V projection)
//   MODE 1: f16 (token,c) output + RoPE    (Q/K projections)
//   MODE 2: fp32 NCHW output + residual    (output projection)
// wave tile: 32(M) x 64(N) (2 A-frags share each B-frag -> 1.5 b128/WMMA),
// K stepped by 32 with v_wmma_f32_16x16x32_f16, 8 accumulators per wave.
// ---------------------------------------------------------------------------
template <int MODE>
__global__ void gemm_kernel(const _Float16* __restrict__ A, const _Float16* __restrict__ W,
                            const float* __restrict__ bias, _Float16* __restrict__ out16,
                            const float* __restrict__ res, float* __restrict__ out32) {
    const int lane = threadIdx.x & 31;
    const int wv   = threadIdx.x >> 5;
    const int m0   = blockIdx.x * 128 + wv * 32;
    const int n0   = blockIdx.y * 64;
    const int kh   = lane >> 4;    // lane half selects K sub-range
    const int ln   = lane & 15;

    const long arow0 = (long)(m0 + ln) * 512;
    const long arow1 = (long)(m0 + 16 + ln) * 512;
    v8f acc[2][4];
    #pragma unroll
    for (int t = 0; t < 2; ++t)
        #pragma unroll
        for (int c = 0; c < 4; ++c) acc[t][c] = v8f_zero();

    for (int k0 = 0; k0 < 512; k0 += 32) {
        AF a0, a1;
        a0.u[0] = *(const uint4*)(A + arow0 + k0 + kh * 8);
        a0.u[1] = *(const uint4*)(A + arow0 + k0 + 16 + kh * 8);
        a1.u[0] = *(const uint4*)(A + arow1 + k0 + kh * 8);
        a1.u[1] = *(const uint4*)(A + arow1 + k0 + 16 + kh * 8);
        #pragma unroll
        for (int c = 0; c < 4; ++c) {
            AF bf;
            const _Float16* wp = W + (long)(n0 + c * 16 + ln) * 512 + k0 + kh * 8;
            bf.u[0] = *(const uint4*)(wp);
            bf.u[1] = *(const uint4*)(wp + 16);
            acc[0][c] = __builtin_amdgcn_wmma_f32_16x16x32_f16(
                false, a0.h, false, bf.h, (short)0, acc[0][c], false, false);
            acc[1][c] = __builtin_amdgcn_wmma_f32_16x16x32_f16(
                false, a1.h, false, bf.h, (short)0, acc[1][c], false, false);
        }
    }

    if (MODE < 2) {
        #pragma unroll
        for (int t = 0; t < 2; ++t) {
            #pragma unroll
            for (int c = 0; c < 4; ++c) {
                int col = n0 + c * 16 + ln;
                float bv = bias[col];
                #pragma unroll
                for (int r = 0; r < 8; ++r) {
                    int m = m0 + 16 * t + r + 8 * kh;   // token index
                    float v = acc[t][c][r] + bv;
                    if (MODE == 1) {            // RoPE (after bias, as in reference)
                        float p = __shfl_xor(v, 1, 32);  // partner of the (even, odd) pair
                        int pos = m & 4095;
                        int d = col & 63;
                        int j2 = d & ~1;        // 2*j
                        float freq = __expf(-(float)j2 * (9.210340372f / 64.f)); // 10000^(-2j/64)
                        float ang = (float)pos * freq;
                        float s, cs;
                        __sincosf(ang, &s, &cs);
                        v = (d & 1) ? (p * s + v * cs) : (v * cs - p * s);
                    }
                    out16[(long)m * 512 + col] = (_Float16)v;
                }
            }
        }
    } else {
        // fp32 NCHW + bias + residual; rows of one accumulator are consecutive n
        const int bb = m0 >> 12;
        #pragma unroll
        for (int t = 0; t < 2; ++t) {
            const int nb = (m0 & 4095) + 16 * t + 8 * kh;
            #pragma unroll
            for (int c = 0; c < 4; ++c) {
                int col = n0 + c * 16 + ln;
                float bv = bias[col];
                long base = (long)bb * 512 * 4096 + (long)col * 4096 + nb;
                float4 r0 = *(const float4*)(res + base);
                float4 r1 = *(const float4*)(res + base + 4);
                float4 o0, o1;
                o0.x = acc[t][c][0] + bv + r0.x;
                o0.y = acc[t][c][1] + bv + r0.y;
                o0.z = acc[t][c][2] + bv + r0.z;
                o0.w = acc[t][c][3] + bv + r0.w;
                o1.x = acc[t][c][4] + bv + r1.x;
                o1.y = acc[t][c][5] + bv + r1.y;
                o1.z = acc[t][c][6] + bv + r1.z;
                o1.w = acc[t][c][7] + bv + r1.w;
                *(float4*)(out32 + base)     = o0;
                *(float4*)(out32 + base + 4) = o1;
            }
        }
    }
}

// ---------------------------------------------------------------------------
// Windowed attention: one block (4 waves) per (head, window).
// S = Q K^T * 1/8 + bias; softmax; O = P V.  All matmuls via WMMA f16.
// ---------------------------------------------------------------------------
__global__ void attn_kernel(const _Float16* __restrict__ Q, const _Float16* __restrict__ K,
                            const _Float16* __restrict__ V, const float* __restrict__ Bias,
                            _Float16* __restrict__ O) {
    __shared__ _Float16 Vt[64 * 72];        // V transposed: [d][tk], stride 72 (16B aligned)
    __shared__ _Float16 P[4 * 16 * 72];     // per-wave probabilities [tq_local][tk]

    const int tid  = threadIdx.x;
    const int lane = tid & 31, wv = tid >> 5;
    const int kh = lane >> 4, ln = lane & 15;

    const int wid = blockIdx.x;
    const int h   = wid >> 9;
    const int win = wid & 511;
    const int b = win >> 6, hn = (win >> 3) & 7, wn = win & 7;
    const long tokbase = (long)b * 4096 + hn * 512 + wn * 8;
    // token(t) = tokbase + (t>>3)*64 + (t&7)

    // ---- phase 1: stage V transposed into LDS (coalesced 64B row reads) ----
    {
        int tk = tid >> 1;
        int d0 = (tid & 1) * 32;
        long vrow = (tokbase + (tk >> 3) * 64 + (tk & 7)) * 512 + h * 64 + d0;
        V32 v;
        v.u[0] = *(const uint4*)(V + vrow);
        v.u[1] = *(const uint4*)(V + vrow + 8);
        v.u[2] = *(const uint4*)(V + vrow + 16);
        v.u[3] = *(const uint4*)(V + vrow + 24);
        #pragma unroll
        for (int i = 0; i < 32; ++i) Vt[(d0 + i) * 72 + tk] = v.h[i];
    }
    __syncthreads();

    // ---- phase 2: S = Q K^T over d=64 (2 k-steps x 4 col tiles) ----
    const int tq = wv * 16 + ln;
    const long qrow = (tokbase + (tq >> 3) * 64 + (tq & 7)) * 512 + h * 64;
    long krow[4];
    #pragma unroll
    for (int c = 0; c < 4; ++c) {
        int tk = c * 16 + ln;
        krow[c] = (tokbase + (tk >> 3) * 64 + (tk & 7)) * 512 + h * 64;
    }
    v8f acc[4] = {v8f_zero(), v8f_zero(), v8f_zero(), v8f_zero()};
    #pragma unroll
    for (int kd = 0; kd < 64; kd += 32) {
        AF a;
        a.u[0] = *(const uint4*)(Q + qrow + kd + kh * 8);
        a.u[1] = *(const uint4*)(Q + qrow + kd + 16 + kh * 8);
        #pragma unroll
        for (int c = 0; c < 4; ++c) {
            AF bf;
            bf.u[0] = *(const uint4*)(K + krow[c] + kd + kh * 8);
            bf.u[1] = *(const uint4*)(K + krow[c] + kd + 16 + kh * 8);
            acc[c] = __builtin_amdgcn_wmma_f32_16x16x32_f16(
                false, a.h, false, bf.h, (short)0, acc[c], false, false);
        }
    }

    // ---- phase 3: scale + relative bias + row softmax ----
    // D-layout: VGPR r, this lane-half holds row tq_r = wv*16 + r + 8*kh, col tk = c*16 + ln.
    // Rows live entirely inside one 16-lane half -> shfl_xor(1,2,4,8) row reductions.
    const float* Bh = Bias + h * 4096;
    #pragma unroll
    for (int r = 0; r < 8; ++r) {
        int tq_r = wv * 16 + r + 8 * kh;
        float mx = -1e30f;
        #pragma unroll
        for (int c = 0; c < 4; ++c) {
            int tk = c * 16 + ln;
            float s = acc[c][r] * 0.125f + Bh[tq_r * 64 + tk];
            acc[c][r] = s;
            mx = fmaxf(mx, s);
        }
        #pragma unroll
        for (int m = 1; m < 16; m <<= 1) mx = fmaxf(mx, __shfl_xor(mx, m, 32));
        float su = 0.f;
        #pragma unroll
        for (int c = 0; c < 4; ++c) {
            float e = __expf(acc[c][r] - mx);
            acc[c][r] = e;
            su += e;
        }
        #pragma unroll
        for (int m = 1; m < 16; m <<= 1) su += __shfl_xor(su, m, 32);
        float inv = 1.f / su;
        #pragma unroll
        for (int c = 0; c < 4; ++c) {
            int tk = c * 16 + ln;
            P[(wv * 16 + r + 8 * kh) * 72 + tk] = (_Float16)(acc[c][r] * inv);
        }
    }

    // ---- phase 4: O = P V  (P re-read from LDS as A-frags; V from Vt) ----
    v8f oacc[4] = {v8f_zero(), v8f_zero(), v8f_zero(), v8f_zero()};
    const _Float16* Pw = P + wv * 16 * 72;
    #pragma unroll
    for (int t0 = 0; t0 < 64; t0 += 32) {
        AF a;
        a.u[0] = *(const uint4*)(Pw + ln * 72 + t0 + kh * 8);
        a.u[1] = *(const uint4*)(Pw + ln * 72 + t0 + 16 + kh * 8);
        #pragma unroll
        for (int c = 0; c < 4; ++c) {
            const _Float16* vp = Vt + (c * 16 + ln) * 72 + t0 + kh * 8;
            AF bf;
            bf.u[0] = *(const uint4*)(vp);
            bf.u[1] = *(const uint4*)(vp + 16);
            oacc[c] = __builtin_amdgcn_wmma_f32_16x16x32_f16(
                false, a.h, false, bf.h, (short)0, oacc[c], false, false);
        }
    }

    // ---- phase 5: store O in (token, c) f16 for the output projection ----
    #pragma unroll
    for (int c = 0; c < 4; ++c) {
        int d = c * 16 + ln;
        #pragma unroll
        for (int r = 0; r < 8; ++r) {
            int tq_r = wv * 16 + r + 8 * kh;
            long orow = (tokbase + (tq_r >> 3) * 64 + (tq_r & 7)) * 512 + h * 64 + d;
            O[orow] = (_Float16)oacc[c][r];
        }
    }
}

// ---------------------------------------------------------------------------
// Host launch. Workspace layout (bytes), total ~227 MB:
//   qln16  f16 (NT*512)      33554432   (reused as attention output Of)
//   kvln16 f16 (NT*512)      33554432
//   qres   f32 (b,c,n)       67108864
//   Wq16/Wk16/Wv16/Wo16      4*524288
//   Qf/Kf/Vf f16             3*33554432
//   Bias   f32 (8*64*64)     131072
// ---------------------------------------------------------------------------
extern "C" void kernel_launch(void* const* d_in, const int* in_sizes, int n_in,
                              void* d_out, int out_size, void* d_ws, size_t ws_size,
                              hipStream_t stream) {
    (void)in_sizes; (void)n_in; (void)out_size; (void)ws_size;

    const float* q    = (const float*)d_in[0];
    const float* kv   = (const float*)d_in[1];
    const float* g_q  = (const float*)d_in[2];
    const float* b_q  = (const float*)d_in[3];
    const float* g_kv = (const float*)d_in[4];
    const float* b_kv = (const float*)d_in[5];
    const float* Wq   = (const float*)d_in[6];
    const float* bq   = (const float*)d_in[7];
    const float* Wk   = (const float*)d_in[8];
    const float* bk   = (const float*)d_in[9];
    const float* Wv   = (const float*)d_in[10];
    const float* bv   = (const float*)d_in[11];
    const float* Wo   = (const float*)d_in[12];
    const float* bo   = (const float*)d_in[13];
    const float* tab  = (const float*)d_in[14];
    float* out = (float*)d_out;

    char* ws = (char*)d_ws;
    _Float16* qln16  = (_Float16*)ws;  ws += 33554432;   // also Of after attention
    _Float16* kvln16 = (_Float16*)ws;  ws += 33554432;
    float*    qres   = (float*)ws;     ws += 67108864;
    _Float16* Wq16   = (_Float16*)ws;  ws += 524288;
    _Float16* Wk16   = (_Float16*)ws;  ws += 524288;
    _Float16* Wv16   = (_Float16*)ws;  ws += 524288;
    _Float16* Wo16   = (_Float16*)ws;  ws += 524288;
    _Float16* Qf     = (_Float16*)ws;  ws += 33554432;
    _Float16* Kf     = (_Float16*)ws;  ws += 33554432;
    _Float16* Vf     = (_Float16*)ws;  ws += 33554432;
    float*    Bias   = (float*)ws;     ws += 131072;
    _Float16* Of     = qln16;          // q-LN f16 is dead after the Q projection

    ln_kernel<<<dim3(2048), dim3(256), 0, stream>>>(q, g_q, b_q, qln16, qres);
    ln_kernel<<<dim3(2048), dim3(256), 0, stream>>>(kv, g_kv, b_kv, kvln16, nullptr);
    wcvt_kernel<<<dim3(1024), dim3(256), 0, stream>>>(Wq, Wk, Wv, Wo, Wq16, Wk16, Wv16, Wo16);
    bias_build_kernel<<<dim3(128), dim3(256), 0, stream>>>(tab, Bias);

    gemm_kernel<1><<<dim3(256, 8), dim3(128), 0, stream>>>(qln16,  Wq16, bq, Qf, nullptr, nullptr);
    gemm_kernel<1><<<dim3(256, 8), dim3(128), 0, stream>>>(kvln16, Wk16, bk, Kf, nullptr, nullptr);
    gemm_kernel<0><<<dim3(256, 8), dim3(128), 0, stream>>>(kvln16, Wv16, bv, Vf, nullptr, nullptr);

    attn_kernel<<<dim3(4096), dim3(128), 0, stream>>>(Qf, Kf, Vf, Bias, Of);

    gemm_kernel<2><<<dim3(256, 8), dim3(128), 0, stream>>>(Of, Wo16, bo, nullptr, qres, out);
}
